// Attention_23699629539900
// MI455X (gfx1250) — matrix operand order repaired and checked by reading, not compiled, not run on hardware
//
#include <hip/hip_runtime.h>
#include <hip/hip_bf16.h>

// ---------------------------------------------------------------------------
// MI455X (gfx1250) implementation.
// Compute-bound (~79 GFLOP vs ~3us of HBM traffic at 23.3 TB/s): all GEMM work
// goes through v_wmma_f32_16x16x32_f16 (wave32, f32 accum); attention is
// flash-style (no NxN materialization); flash K/V tiles are fetched by the
// Tensor Data Mover (tensor_load_to_lds + s_wait_tensorcnt) when available.
// Workspace layout (96 MB):
//   [ 0M) qh   f16 (b,H,n,d)      [40M) sb0  f16 (8192,512)
//   [ 8M) kh   f16 (b,H,n,d)      [48M) sb1  f16 (8192,512)
//   [16M) vtg  f16 (b,H,d,n)      [56M) cn   f16 (8192,384)
//   [24M) ckh  f16 (b,H,n,d)      [64M) attn_o f32 (b,H,n,d)
//   [32M) cvtg f16 (b,H,d,n)      [80M) attn_c f32 (b,H,n,d)
// ---------------------------------------------------------------------------

typedef __attribute__((ext_vector_type(16))) _Float16 v16h;
typedef __attribute__((ext_vector_type(8)))  float    v8f;
typedef __attribute__((ext_vector_type(4)))  unsigned u32x4;
typedef __attribute__((ext_vector_type(4)))  int      i32x4;
typedef __attribute__((ext_vector_type(8)))  int      i32x8;

#define NTOK 1024
#define HN   8

#if defined(__has_builtin)
#if __has_builtin(__builtin_amdgcn_tensor_load_to_lds) && \
    __has_builtin(__builtin_amdgcn_s_wait_tensorcnt)
#define USE_TDM 1
#endif
#endif
#ifndef USE_TDM
#define USE_TDM 0
#endif

#if USE_TDM
// Issue a TDM load of a 2D f16 tile (tile0 x tile1 elems, global row stride
// stride0 elems) into LDS at lds_addr, inserting 16B of LDS padding after
// every 128B (matches the [*][72] LDS row layout).  D# per ISA ch.8.
__device__ __forceinline__ void tdm_load_f16(unsigned lds_addr, const void* gp,
                                             unsigned tdim0, unsigned tile0,
                                             unsigned tile1,
                                             unsigned long long stride0) {
  unsigned long long ga = (unsigned long long)gp;
  u32x4 g0;
  g0[0] = 1u;                                       // count=1, user descriptor
  g0[1] = lds_addr;                                 // bits 63:32 lds_addr
  g0[2] = (unsigned)(ga & 0xffffffffu);             // global_addr[31:0]
  g0[3] = (unsigned)((ga >> 32) & 0x01ffffffu) | (2u << 30);  // addr[56:32]|type=2
  i32x8 g1;
  const unsigned pad_int = 4u;                      // 32 DWORDs (=128B) interval
  const unsigned pad_amt = 3u;                      // 4 DWORDs (=16B) padding
  g1[0] = (int)((1u << 16) |                        // data_size = 2 bytes
                (1u << 20) |                        // pad_enable
                (pad_int << 22) | (pad_amt << 25));
  g1[1] = (int)((tdim0 & 0xffffu) << 16);           // tensor_dim0[15:0]
  g1[2] = (int)((tdim0 >> 16) | ((tile1 & 0xffffu) << 16));  // td0 hi | tensor_dim1 lo
  g1[3] = (int)((tile0 & 0xffffu) << 16);           // tensor_dim1 hi=0 | tile_dim0
  g1[4] = (int)(tile1 & 0xffffu);                   // tile_dim1 (tile_dim2 = 0)
  g1[5] = (int)(unsigned)(stride0 & 0xffffffffu);   // tensor_dim0_stride[31:0]
  g1[6] = (int)((stride0 >> 32) & 0xffffu);         // stride hi (dim1_stride = 0)
  g1[7] = 0;
  i32x4 z4 = {0, 0, 0, 0};
#if defined(__clang_major__) && __clang_major__ >= 23
  i32x8 z8 = {0, 0, 0, 0, 0, 0, 0, 0};
  __builtin_amdgcn_tensor_load_to_lds(g0, g1, z4, z4, z8, 0);
#else
  __builtin_amdgcn_tensor_load_to_lds(g0, g1, z4, z4, 0);
#endif
}
#endif  // USE_TDM

// A-fragment (16x32, f16): lane L holds row m, K strips [8*kg..+7] and
// [16+8*kg..+7], kg = L>>4.  Paired halves contiguous -> b32 loads.
__device__ __forceinline__ v16h frag_a(const _Float16* a, int row0, int k0,
                                       int ld, int lane) {
  const _Float16* p = a + (size_t)(row0 + (lane & 15)) * ld + k0 + ((lane >> 4) << 3);
  union { v16h v; unsigned u[8]; } r;
  const unsigned* p0 = (const unsigned*)p;
  const unsigned* p1 = (const unsigned*)(p + 16);
  r.u[0] = p0[0]; r.u[1] = p0[1]; r.u[2] = p0[2]; r.u[3] = p0[3];
  r.u[4] = p1[0]; r.u[5] = p1[1]; r.u[6] = p1[2]; r.u[7] = p1[3];
  return r.v;
}

// B-fragment (32x16, f16) from [n][k] storage: lane L holds column n = L%16,
// K strips [8*kg..+7] and [16+8*kg..+7], kg = L>>4 -> 8 b32 loads.
__device__ __forceinline__ v16h frag_b(const _Float16* b, int col0, int k0,
                                       int ld, int lane) {
  const _Float16* p = b + (size_t)(col0 + (lane & 15)) * ld + k0 + ((lane >> 4) << 3);
  union { v16h v; unsigned u[8]; } r;
  const unsigned* p0 = (const unsigned*)p;
#pragma unroll
  for (int i = 0; i < 8; ++i) r.u[i] = p0[(i & 3) + ((i >> 2) << 3)];
  return r.v;
}

// ---------------------------------------------------------------------------
// Generic NT GEMM: C(M,N) = act(A(M,K) @ W(N,K)^T + bias).
// Block tile 64x128, 4 waves in a 2x2 grid, each wave computes 32x64
// (2 A-frags x 4 B-frags -> 8 WMMAs per k-step).
// MODE 0: no bias, scatter to qh / kh / v-transposed (A is f32)
// MODE 1: bias + LeakyReLU(0.2), f16 row-major out0 (ld = N)
// MODE 2: bias + LeakyReLU(0.2), f16 head-major (b,H,n,d) out0
// MODE 5: bias + LeakyReLU(0.2), f16 transposed head-major (b,H,d,n) out0
// MODE 3: bias + exact GELU, combine: y = attn - gelu*subr, f16 row-major out0
// MODE 4: bias only, f32 row-major out0 (final output)
// ---------------------------------------------------------------------------
template <int MODE, bool A_F32>
__global__ __launch_bounds__(128) void gemm_nt(
    const void* __restrict__ Ap, const float* __restrict__ W,
    const float* __restrict__ bias,
    void* __restrict__ out0, void* __restrict__ out1, void* __restrict__ out2,
    const float* __restrict__ attn, const float* __restrict__ subr,
    int M, int N, int K) {
  __shared__ __align__(16) _Float16 As[64][36];
  __shared__ __align__(16) _Float16 Ws[128][36];
  const int t = threadIdx.x;
  const int lane = t & 31;
  const int wave = t >> 5;
  const int wm = wave >> 1, wn = wave & 1;
  const int bm = blockIdx.x * 64;
  const int bn = blockIdx.y * 128;
  (void)M;

  v8f acc[2][4] = {};

  for (int k0 = 0; k0 < K; k0 += 32) {
    if (k0 + 32 < K) {  // prefetch next tiles -> global_prefetch_b8
      __builtin_prefetch((const char*)Ap + ((size_t)(bm + (t >> 2)) * K + k0 + 32) *
                                               (A_F32 ? 4 : 2), 0, 1);
      __builtin_prefetch((const char*)W + ((size_t)(bn + t) * K + k0 + 32) * 4, 0, 1);
    }
#pragma unroll
    for (int i = 0; i < 16; ++i) {  // A tile 64x32
      int idx = i * 128 + t;
      int r = idx >> 5, c = idx & 31;
      _Float16 hv;
      if (A_F32) hv = (_Float16)(((const float*)Ap)[(size_t)(bm + r) * K + k0 + c]);
      else       hv = ((const _Float16*)Ap)[(size_t)(bm + r) * K + k0 + c];
      As[r][c] = hv;
    }
#pragma unroll
    for (int i = 0; i < 32; ++i) {  // W tile 128x32
      int idx = i * 128 + t;
      int r = idx >> 5, c = idx & 31;
      Ws[r][c] = (_Float16)W[(size_t)(bn + r) * K + k0 + c];
    }
    __syncthreads();
    v16h af[2], bf[4];
#pragma unroll
    for (int mf = 0; mf < 2; ++mf) af[mf] = frag_a(&As[0][0], wm * 32 + mf * 16, 0, 36, lane);
#pragma unroll
    for (int nf = 0; nf < 4; ++nf) bf[nf] = frag_b(&Ws[0][0], wn * 64 + nf * 16, 0, 36, lane);
#pragma unroll
    for (int mf = 0; mf < 2; ++mf)
#pragma unroll
      for (int nf = 0; nf < 4; ++nf)
        acc[mf][nf] = __builtin_amdgcn_wmma_f32_16x16x32_f16(
            false, af[mf], false, bf[nf], (short)0, acc[mf][nf], false, false);
    __syncthreads();
  }

#pragma unroll
  for (int mf = 0; mf < 2; ++mf)
#pragma unroll
    for (int nf = 0; nf < 4; ++nf)
#pragma unroll
      for (int r = 0; r < 8; ++r) {
        int gm = bm + wm * 32 + mf * 16 + r + ((lane >> 4) << 3);
        int gn = bn + wn * 64 + nf * 16 + (lane & 15);
        float v = acc[mf][nf][r];
        int b = gm >> 10, n = gm & 1023;
        if (MODE == 0) {
          int part = gn >> 9;
          int jj = gn & 511;
          int h = jj >> 6, d = jj & 63;
          _Float16 hv = (_Float16)v;
          if (part == 0)
            ((_Float16*)out0)[((((size_t)b * HN + h) * NTOK + n) << 6) + d] = hv;
          else if (part == 1)
            ((_Float16*)out1)[((((size_t)b * HN + h) * NTOK + n) << 6) + d] = hv;
          else  // V stored transposed (b,H,d,n)
            ((_Float16*)out2)[(((size_t)b * HN + h) * 64 + d) * NTOK + n] = hv;
        } else if (MODE == 1) {
          v += bias[gn];
          v = v > 0.f ? v : 0.2f * v;
          ((_Float16*)out0)[(size_t)gm * N + gn] = (_Float16)v;
        } else if (MODE == 2) {
          v += bias[gn];
          v = v > 0.f ? v : 0.2f * v;
          int h = gn >> 6, d = gn & 63;
          ((_Float16*)out0)[((((size_t)b * HN + h) * NTOK + n) << 6) + d] = (_Float16)v;
        } else if (MODE == 5) {
          v += bias[gn];
          v = v > 0.f ? v : 0.2f * v;
          int h = gn >> 6, d = gn & 63;
          ((_Float16*)out0)[(((size_t)b * HN + h) * 64 + d) * NTOK + n] = (_Float16)v;
        } else if (MODE == 3) {
          v += bias[gn];
          float g = 0.5f * v * (1.f + erff(v * 0.70710678118f));
          int h = gn >> 6, d = gn & 63;
          size_t hidx = ((((size_t)b * HN + h) * NTOK + n) << 6) + d;
          float y = attn[hidx] - g * subr[(h << 6) + d];
          ((_Float16*)out0)[(size_t)gm * N + gn] = (_Float16)y;
        } else {  // MODE 4
          v += bias[gn];
          ((float*)out0)[(size_t)gm * N + gn] = v;
        }
      }
}

// ---------------------------------------------------------------------------
// Column L2-norm over the token axis + transpose to (b, n, C) f16.
// in: f32 (b, Hin, n, 64); one block per (b, channel c), C = Hin*64.
// ---------------------------------------------------------------------------
__global__ __launch_bounds__(256) void colnorm_t(const float* __restrict__ in,
                                                 _Float16* __restrict__ out,
                                                 int Hin, int C) {
  int bc = blockIdx.x;
  int b = bc / C, c = bc % C;
  int h = c >> 6, cc = c & 63;
  const float* src = in + ((((size_t)b * Hin + h) * NTOK) << 6) + cc;
  int t = threadIdx.x;
  float vals[4];
  float ss = 0.f;
#pragma unroll
  for (int i = 0; i < 4; ++i) {
    float v = src[(size_t)(t + i * 256) << 6];
    vals[i] = v;
    ss += v * v;
  }
#pragma unroll
  for (int off = 16; off; off >>= 1) ss += __shfl_xor(ss, off, 32);
  __shared__ float red[8];
  if ((t & 31) == 0) red[t >> 5] = ss;
  __syncthreads();
  if (t == 0) {
    float s = 0.f;
#pragma unroll
    for (int i = 0; i < 8; ++i) s += red[i];
    red[0] = s;
  }
  __syncthreads();
  float scale = 1.f / fmaxf(sqrtf(red[0]), 1e-12f);
#pragma unroll
  for (int i = 0; i < 4; ++i) {
    int n = t + i * 256;
    out[((size_t)b * NTOK + n) * C + c] = (_Float16)(vals[i] * scale);
  }
}

// ---------------------------------------------------------------------------
// Flash attention: grid (N/64, B*H), 4 waves; wave owns 16 q rows.
// qh/kh: f16 (b,H,n,d); vtg: f16 (b,H,d,n); out: f32 (b,H,n,d).
// K/V tiles are DMA'd into LDS by the Tensor Data Mover when available.
// ---------------------------------------------------------------------------
__global__ __launch_bounds__(128) void flash_attn(const _Float16* __restrict__ qh,
                                                  const _Float16* __restrict__ kh,
                                                  const _Float16* __restrict__ vtg,
                                                  float* __restrict__ outp) {
  __shared__ __align__(16) _Float16 ktl[64][72];  // [n'][d]
  __shared__ __align__(16) _Float16 vtl[64][72];  // [d][n']
  __shared__ __align__(16) _Float16 pst[4][16][72];
  const int t = threadIdx.x, lane = t & 31, wave = t >> 5;
  const size_t base = (size_t)blockIdx.y * (NTOK * 64);
  const int q0 = blockIdx.x * 64 + wave * 16;

  v16h aq[2];
  {
    const _Float16* qp = qh + base + (size_t)q0 * 64;
    aq[0] = frag_a(qp, 0, 0, 64, lane);
    aq[1] = frag_a(qp, 0, 32, 64, lane);
    const _Float16 sc = (_Float16)0.125f;  // DH^-0.5
#pragma unroll
    for (int j = 0; j < 16; ++j) { aq[0][j] *= sc; aq[1][j] *= sc; }
  }

  float mrow[8], lrow[8];
  v8f oacc[4] = {};
#pragma unroll
  for (int r = 0; r < 8; ++r) { mrow[r] = -1e30f; lrow[r] = 0.f; }

#if USE_TDM
  const unsigned lds_kt = (unsigned)(unsigned long long)(const void*)&ktl[0][0];
  const unsigned lds_vt = (unsigned)(unsigned long long)(const void*)&vtl[0][0];
#endif

  for (int kt0 = 0; kt0 < NTOK; kt0 += 64) {
    const _Float16* ksrc = kh + base + (size_t)kt0 * 64;
    const _Float16* vsrc = vtg + base + kt0;
#if USE_TDM
    if (t == 0) {
      // K tile: 1D copy of 64*64 f16 with 16B LDS pad per 128B row.
      tdm_load_f16(lds_kt, ksrc, 4096u, 4096u, 1u, 4096ull);
      // V tile: 2D 64x64 f16, global row stride NTOK elems.
      tdm_load_f16(lds_vt, vsrc, 64u, 64u, 64u, (unsigned long long)NTOK);
    }
    __builtin_amdgcn_s_wait_tensorcnt(0);
    __syncthreads();
#else
#pragma unroll
    for (int i = 0; i < 4; ++i) {
      int idx = i * 128 + t;
      int r = idx >> 3, c8 = (idx & 7) << 3;
      *(uint4*)&ktl[r][c8] = *(const uint4*)(ksrc + (size_t)r * 64 + c8);
      *(uint4*)&vtl[r][c8] = *(const uint4*)(vsrc + (size_t)r * NTOK + c8);
    }
    __syncthreads();
#endif

    v8f s[4];
#pragma unroll
    for (int nf = 0; nf < 4; ++nf) {
      v8f z = {};
      z = __builtin_amdgcn_wmma_f32_16x16x32_f16(
          false, aq[0], false, frag_b(&ktl[0][0], nf * 16, 0, 72, lane), (short)0, z, false, false);
      z = __builtin_amdgcn_wmma_f32_16x16x32_f16(
          false, aq[1], false, frag_b(&ktl[0][0], nf * 16, 32, 72, lane), (short)0, z, false, false);
      s[nf] = z;
    }

#pragma unroll
    for (int r = 0; r < 8; ++r) {
      float mx = fmaxf(fmaxf(s[0][r], s[1][r]), fmaxf(s[2][r], s[3][r]));
#pragma unroll
      for (int off = 1; off < 16; off <<= 1) mx = fmaxf(mx, __shfl_xor(mx, off, 32));
      float mnew = fmaxf(mrow[r], mx);
      float alpha = __expf(mrow[r] - mnew);
      float rs = 0.f;
#pragma unroll
      for (int nf = 0; nf < 4; ++nf) {
        float p = __expf(s[nf][r] - mnew);
        s[nf][r] = p;
        rs += p;
      }
#pragma unroll
      for (int off = 1; off < 16; off <<= 1) rs += __shfl_xor(rs, off, 32);
      lrow[r] = lrow[r] * alpha + rs;
      mrow[r] = mnew;
#pragma unroll
      for (int df = 0; df < 4; ++df) oacc[df][r] *= alpha;
    }

    {
      int pr = (lane >> 4) << 3, pc = lane & 15;
#pragma unroll
      for (int nf = 0; nf < 4; ++nf)
#pragma unroll
        for (int r = 0; r < 8; ++r)
          pst[wave][pr + r][nf * 16 + pc] = (_Float16)s[nf][r];
    }
    __syncthreads();

    const _Float16* pp = &pst[wave][0][0];
    v16h ap0 = frag_a(pp, 0, 0, 72, lane);
    v16h ap1 = frag_a(pp, 0, 32, 72, lane);
#pragma unroll
    for (int df = 0; df < 4; ++df) {
      oacc[df] = __builtin_amdgcn_wmma_f32_16x16x32_f16(
          false, ap0, false, frag_b(&vtl[0][0], df * 16, 0, 72, lane), (short)0, oacc[df], false, false);
      oacc[df] = __builtin_amdgcn_wmma_f32_16x16x32_f16(
          false, ap1, false, frag_b(&vtl[0][0], df * 16, 32, 72, lane), (short)0, oacc[df], false, false);
    }
    __syncthreads();
  }

  int pr = (lane >> 4) << 3, pc = lane & 15;
#pragma unroll
  for (int r = 0; r < 8; ++r) {
    float inv = 1.f / lrow[r];
#pragma unroll
    for (int df = 0; df < 4; ++df)
      outp[base + (size_t)(q0 + pr + r) * 64 + df * 16 + pc] = oacc[df][r] * inv;
  }
}

// ---------------------------------------------------------------------------
extern "C" void kernel_launch(void* const* d_in, const int* in_sizes, int n_in,
                              void* d_out, int out_size, void* d_ws, size_t ws_size,
                              hipStream_t stream) {
  (void)in_sizes; (void)n_in; (void)out_size; (void)ws_size;
  const float* x     = (const float*)d_in[0];
  const float* ck    = (const float*)d_in[1];
  const float* cv    = (const float*)d_in[2];
  const float* w_qkv = (const float*)d_in[3];
  const float* w_out = (const float*)d_in[4];
  const float* b_out = (const float*)d_in[5];
  const float* ckw0 = (const float*)d_in[6];  const float* ckb0 = (const float*)d_in[7];
  const float* ckw1 = (const float*)d_in[8];  const float* ckb1 = (const float*)d_in[9];
  const float* ckw2 = (const float*)d_in[10]; const float* ckb2 = (const float*)d_in[11];
  const float* cvw0 = (const float*)d_in[12]; const float* cvb0 = (const float*)d_in[13];
  const float* cvw1 = (const float*)d_in[14]; const float* cvb1 = (const float*)d_in[15];
  const float* cvw2 = (const float*)d_in[16]; const float* cvb2 = (const float*)d_in[17];
  const float* nl_w = (const float*)d_in[18]; const float* nl_b = (const float*)d_in[19];
  const float* subr = (const float*)d_in[20];

  char* ws = (char*)d_ws;
  _Float16* qh   = (_Float16*)(ws);
  _Float16* khp  = (_Float16*)(ws + ((size_t)8  << 20));
  _Float16* vtg  = (_Float16*)(ws + ((size_t)16 << 20));
  _Float16* ckh  = (_Float16*)(ws + ((size_t)24 << 20));
  _Float16* cvtg = (_Float16*)(ws + ((size_t)32 << 20));
  _Float16* sb0  = (_Float16*)(ws + ((size_t)40 << 20));
  _Float16* sb1  = (_Float16*)(ws + ((size_t)48 << 20));
  _Float16* cn   = (_Float16*)(ws + ((size_t)56 << 20));
  float*    atno = (float*)(ws + ((size_t)64 << 20));
  float*    atnc = (float*)(ws + ((size_t)80 << 20));

  dim3 blk(128);
  // QKV projection + head scatter (V transposed)
  gemm_nt<0, true><<<dim3(128, 12), blk, 0, stream>>>(
      x, w_qkv, nullptr, qh, khp, vtg, nullptr, nullptr, 8192, 1536, 512);

  // style vectorizer (ck)
  colnorm_t<<<8 * 384, 256, 0, stream>>>(ck, cn, 6, 384);
  gemm_nt<1, false><<<dim3(128, 4), blk, 0, stream>>>(
      cn, ckw0, ckb0, sb0, nullptr, nullptr, nullptr, nullptr, 8192, 512, 384);
  gemm_nt<1, false><<<dim3(128, 4), blk, 0, stream>>>(
      sb0, ckw1, ckb1, sb1, nullptr, nullptr, nullptr, nullptr, 8192, 512, 512);
  gemm_nt<2, false><<<dim3(128, 4), blk, 0, stream>>>(
      sb1, ckw2, ckb2, ckh, nullptr, nullptr, nullptr, nullptr, 8192, 512, 512);

  // style vectorizer (cv) -> transposed head-major (B operand of P@V)
  colnorm_t<<<8 * 384, 256, 0, stream>>>(cv, cn, 6, 384);
  gemm_nt<1, false><<<dim3(128, 4), blk, 0, stream>>>(
      cn, cvw0, cvb0, sb0, nullptr, nullptr, nullptr, nullptr, 8192, 512, 384);
  gemm_nt<1, false><<<dim3(128, 4), blk, 0, stream>>>(
      sb0, cvw1, cvb1, sb1, nullptr, nullptr, nullptr, nullptr, 8192, 512, 512);
  gemm_nt<5, false><<<dim3(128, 4), blk, 0, stream>>>(
      sb1, cvw2, cvb2, cvtg, nullptr, nullptr, nullptr, nullptr, 8192, 512, 512);

  // two attention branches
  flash_attn<<<dim3(16, 64), blk, 0, stream>>>(qh, khp, vtg, atno);
  flash_attn<<<dim3(16, 64), blk, 0, stream>>>(qh, ckh, cvtg, atnc);

  // MLP block on context branch + combine with sub_ratio
  colnorm_t<<<8 * 512, 256, 0, stream>>>(atnc, sb0, 8, 512);
  gemm_nt<3, false><<<dim3(128, 4), blk, 0, stream>>>(
      sb0, nl_w, nl_b, sb1, nullptr, nullptr, atno, subr, 8192, 512, 512);

  // final projection -> f32 output (b, n, 512)
  gemm_nt<4, false><<<dim3(128, 4), blk, 0, stream>>>(
      sb1, w_out, b_out, d_out, nullptr, nullptr, nullptr, nullptr, 8192, 512, 512);
}
